// TransformerEncoderLayer_76802605187350
// MI455X (gfx1250) — matrix operand order repaired and checked by reading, not compiled, not run on hardware
//
#include <hip/hip_runtime.h>
#include <hip/hip_bf16.h>

typedef __bf16 bf16;
typedef float v8f   __attribute__((ext_vector_type(8)));
typedef bf16  bf16x16 __attribute__((ext_vector_type(16)));

union FragAB {
    bf16x16 v;
    uint4   u[2];
    bf16    e[16];
};

union Pack8 {
    uint4 u;
    bf16  e[8];
};

__device__ __forceinline__ v8f zero8() {
    v8f z;
    #pragma unroll
    for (int i = 0; i < 8; ++i) z[i] = 0.0f;
    return z;
}

// ---------------------------------------------------------------------------
// CDNA5 async global->LDS copy (ASYNCcnt path).
// Low 32 bits of a generic pointer to __shared__ == LDS byte offset
// ---------------------------------------------------------------------------
__device__ __forceinline__ void async_copy16(const void* gptr, void* lptr) {
    unsigned lds = (unsigned)(unsigned long long)(uintptr_t)lptr;
    unsigned long long ga = (unsigned long long)(uintptr_t)gptr;
    asm volatile("global_load_async_to_lds_b128 %0, %1, off"
                 :: "v"(lds), "v"(ga) : "memory");
}

__device__ __forceinline__ void wait_async0() {
    asm volatile("s_wait_asynccnt 0x0" ::: "memory");
}

// ---------------------------------------------------------------------------
// Conversion kernels
// ---------------------------------------------------------------------------
__global__ __launch_bounds__(256) void cvt_f32_bf16(const float* __restrict__ in,
                                                    bf16* __restrict__ out, int n) {
    int i = blockIdx.x * 256 + threadIdx.x;
    if (i < n) out[i] = (bf16)in[i];
}

// in [K][N] fp32 row-major  ->  out [N][K] bf16 row-major (transposed)
__global__ __launch_bounds__(256) void transpose_cvt(const float* __restrict__ in,
                                                     bf16* __restrict__ out,
                                                     int K, int N) {
    int i = blockIdx.x * 256 + threadIdx.x;
    if (i < K * N) {
        int k = i / N, n = i - k * N;
        out[(size_t)n * K + k] = (bf16)in[i];
    }
}

// Wq [H][D][DK] fp32 -> out[(h*DK+kk)][d] bf16  (i.e. [N=512][K=512], transposed)
__global__ __launch_bounds__(256) void cvt_qkv_w(const float* __restrict__ in,
                                                 bf16* __restrict__ out) {
    const int D = 512, DK = 64;
    int i = blockIdx.x * 256 + threadIdx.x; // over 512*512
    int h  = i / (D * DK);
    int r  = i - h * D * DK;
    int d  = r / DK;
    int kk = r - d * DK;
    out[(size_t)(h * DK + kk) * D + d] = (bf16)in[i];
}

// ---------------------------------------------------------------------------
// Tiled bf16 GEMM with WMMA:  C[M][N] = A[M][K] * Bt[N][K]^T + bias
// Double-buffered LDS with async global->LDS staging.
// MODE 0: outB[M][N] = bf16(acc + bias[n])
// MODE 1: outF[M][N] = acc + bias[n] + resid[M][N]   (fp32)
// ---------------------------------------------------------------------------
#define GBM 128
#define GBN 128
#define GBK 64

__device__ __forceinline__ void stage_tiles_async(
    const bf16* __restrict__ A, const bf16* __restrict__ Bt,
    bf16 (*Alds)[GBK], bf16 (*Blds)[GBK],
    int bm0, int bn0, int k0, int Kdim, int tid) {
    // A tile: 128x64 bf16 = 1024 16B chunks / 256 threads
    for (int c = tid; c < (GBM * GBK / 8); c += 256) {
        int r = c >> 3, cc = (c & 7) * 8;
        async_copy16(&A[(size_t)(bm0 + r) * Kdim + k0 + cc], &Alds[r][cc]);
    }
    for (int c = tid; c < (GBN * GBK / 8); c += 256) {
        int r = c >> 3, cc = (c & 7) * 8;
        async_copy16(&Bt[(size_t)(bn0 + r) * Kdim + k0 + cc], &Blds[r][cc]);
    }
}

template <int MODE>
__global__ __launch_bounds__(256)
void gemm_bf16(const bf16* __restrict__ A, const bf16* __restrict__ Bt,
               const float* __restrict__ bias, const float* __restrict__ resid,
               float* __restrict__ outF, bf16* __restrict__ outB,
               int Mdim, int Ndim, int Kdim) {
    __shared__ bf16 Alds[2][GBM][GBK];
    __shared__ bf16 Blds[2][GBN][GBK];

    const int tid  = threadIdx.x;
    const int lane = tid & 31;
    const int wave = tid >> 5;     // 0..7
    const int wm   = wave >> 1;    // 0..3 (M direction)
    const int wn   = wave & 1;     // 0..1 (N direction)
    const int l16  = lane & 15;
    const int half = lane >> 4;
    const int bm0  = blockIdx.y * GBM;
    const int bn0  = blockIdx.x * GBN;

    v8f acc[2][4];
    #pragma unroll
    for (int i = 0; i < 2; ++i)
        #pragma unroll
        for (int j = 0; j < 4; ++j) acc[i][j] = zero8();

    const int nk = Kdim / GBK;

    // prologue: stage tile 0
    stage_tiles_async(A, Bt, Alds[0], Blds[0], bm0, bn0, 0, Kdim, tid);
    wait_async0();
    __syncthreads();

    for (int it = 0; it < nk; ++it) {
        const int cur = it & 1;
        // kick off next tile while computing this one
        if (it + 1 < nk)
            stage_tiles_async(A, Bt, Alds[cur ^ 1], Blds[cur ^ 1],
                              bm0, bn0, (it + 1) * GBK, Kdim, tid);

        #pragma unroll
        for (int ks = 0; ks < 2; ++ks) {
            FragAB af[2], bfv[4];
            #pragma unroll
            for (int i = 0; i < 2; ++i) {
                int row = wm * 32 + i * 16 + l16;
                int kc  = ks * 32 + half * 8;   // A layout: K 0-7/8-15 then +16
                af[i].u[0] = *(const uint4*)&Alds[cur][row][kc];
                af[i].u[1] = *(const uint4*)&Alds[cur][row][kc + 16];
            }
            #pragma unroll
            for (int j = 0; j < 4; ++j) {
                int col = wn * 64 + j * 16 + l16;
                int kc  = ks * 32 + half * 8;   // B layout: K 0-7/8-15 then +16
                bfv[j].u[0] = *(const uint4*)&Blds[cur][col][kc];
                bfv[j].u[1] = *(const uint4*)&Blds[cur][col][kc + 16];
            }
            #pragma unroll
            for (int i = 0; i < 2; ++i)
                #pragma unroll
                for (int j = 0; j < 4; ++j)
                    acc[i][j] = __builtin_amdgcn_wmma_f32_16x16x32_bf16(
                        false, af[i].v, false, bfv[j].v, (short)0, acc[i][j],
                        false, false);
        }

        wait_async0();        // my next-tile asyncs have landed
        __syncthreads();      // everyone's have landed; cur buffer free
    }

    // epilogue
    #pragma unroll
    for (int j = 0; j < 4; ++j) {
        int col = bn0 + wn * 64 + j * 16 + l16;
        float bv = bias[col];
        #pragma unroll
        for (int i = 0; i < 2; ++i) {
            #pragma unroll
            for (int r = 0; r < 8; ++r) {
                int row = bm0 + wm * 32 + i * 16 + r + half * 8;
                size_t idx = (size_t)row * Ndim + col;
                float v = acc[i][j][r] + bv;
                if (MODE == 0) {
                    outB[idx] = (bf16)v;
                } else {
                    outF[idx] = v + resid[idx];
                }
            }
        }
    }
}

// ---------------------------------------------------------------------------
// Flash-style attention: one block = (b, h, 128 queries); 8 waves x 16 queries
// ---------------------------------------------------------------------------
__global__ __launch_bounds__(256)
void attn_kernel(const bf16* __restrict__ Qb, const bf16* __restrict__ Kb,
                 const bf16* __restrict__ Vb, bf16* __restrict__ Ctxb) {
    const int S = 1024, D = 512;
    __shared__ bf16 Klds[32][64];       // [key][dk]
    __shared__ bf16 Vlds[64][32];       // [vdim][key] (transposed)
    __shared__ bf16 Plds[8][16][32];    // per-wave P scratch [m][key]

    const int b  = blockIdx.z;
    const int h  = blockIdx.y;
    const int q0 = blockIdx.x * 128;

    const int tid  = threadIdx.x;
    const int lane = tid & 31;
    const int wave = tid >> 5;
    const int l16  = lane & 15;
    const int half = lane >> 4;

    // Load the Q fragments for this wave's 16 queries (K = 64 -> 2 frags)
    const int qtok = b * S + q0 + wave * 16 + l16;
    const bf16* qrow = Qb + (size_t)qtok * D + h * 64;
    FragAB qf[2];
    #pragma unroll
    for (int ks = 0; ks < 2; ++ks) {
        int kc = ks * 32 + half * 8;
        qf[ks].u[0] = *(const uint4*)&qrow[kc];
        qf[ks].u[1] = *(const uint4*)&qrow[kc + 16];
    }

    v8f acc[4];
    #pragma unroll
    for (int j = 0; j < 4; ++j) acc[j] = zero8();
    float mrun[8], lrun[8];
    #pragma unroll
    for (int r = 0; r < 8; ++r) { mrun[r] = -1e30f; lrun[r] = 0.0f; }

    for (int j0 = 0; j0 < S; j0 += 32) {
        // stage K tile [32][64] via async path (256 chunks -> 1 per thread)
        {
            int r = tid >> 3, cc = (tid & 7) * 8;
            async_copy16(&Kb[(size_t)(b * S + j0 + r) * D + h * 64 + cc],
                         &Klds[r][cc]);
        }
        // stage V tile transposed: Vlds[v][key] (register round-trip scatter)
        {
            int r = tid >> 3, cc = (tid & 7) * 8;
            Pack8 tv;
            tv.u = *(const uint4*)&Vb[(size_t)(b * S + j0 + r) * D + h * 64 + cc];
            #pragma unroll
            for (int jj = 0; jj < 8; ++jj) Vlds[cc + jj][r] = tv.e[jj];
        }
        wait_async0();
        __syncthreads();

        // scores: 16 queries x 32 keys = two 16x16 C tiles
        v8f s0 = zero8(), s1 = zero8();
        #pragma unroll
        for (int ks = 0; ks < 2; ++ks) {
            FragAB kf0, kf1;
            int kc = ks * 32 + half * 8;
            kf0.u[0] = *(const uint4*)&Klds[l16][kc];
            kf0.u[1] = *(const uint4*)&Klds[l16][kc + 16];
            kf1.u[0] = *(const uint4*)&Klds[16 + l16][kc];
            kf1.u[1] = *(const uint4*)&Klds[16 + l16][kc + 16];
            s0 = __builtin_amdgcn_wmma_f32_16x16x32_bf16(false, qf[ks].v, false,
                                                         kf0.v, (short)0, s0,
                                                         false, false);
            s1 = __builtin_amdgcn_wmma_f32_16x16x32_bf16(false, qf[ks].v, false,
                                                         kf1.v, (short)0, s1,
                                                         false, false);
        }

        // online softmax update (rows live per-register; cols across 16 lanes)
        #pragma unroll
        for (int r = 0; r < 8; ++r) {
            float a0 = s0[r] * 0.125f;      // 1/sqrt(64)
            float a1 = s1[r] * 0.125f;
            float mx = fmaxf(a0, a1);
            #pragma unroll
            for (int off = 1; off < 16; off <<= 1)
                mx = fmaxf(mx, __shfl_xor(mx, off));
            float mnew = fmaxf(mrun[r], mx);
            float f    = __expf(mrun[r] - mnew);
            mrun[r]    = mnew;
            float p0 = __expf(a0 - mnew);
            float p1 = __expf(a1 - mnew);
            float rs = p0 + p1;
            #pragma unroll
            for (int off = 1; off < 16; off <<= 1)
                rs += __shfl_xor(rs, off);
            lrun[r] = lrun[r] * f + rs;
            // write P (C-layout -> LDS, re-read below in A-layout)
            int m = r + 8 * half;
            Plds[wave][m][l16]      = (bf16)p0;
            Plds[wave][m][16 + l16] = (bf16)p1;
            #pragma unroll
            for (int j = 0; j < 4; ++j) acc[j][r] *= f;
        }

        // PV: A = P (16x32), B = V tile (32 keys x 64 vdims -> 4 N tiles)
        FragAB pf;
        {
            int kc = half * 8;
            pf.u[0] = *(const uint4*)&Plds[wave][l16][kc];
            pf.u[1] = *(const uint4*)&Plds[wave][l16][kc + 16];
        }
        #pragma unroll
        for (int j = 0; j < 4; ++j) {
            FragAB vf;
            int vcol = j * 16 + l16;
            int kc   = half * 8;
            vf.u[0] = *(const uint4*)&Vlds[vcol][kc];
            vf.u[1] = *(const uint4*)&Vlds[vcol][kc + 16];
            acc[j] = __builtin_amdgcn_wmma_f32_16x16x32_bf16(
                false, pf.v, false, vf.v, (short)0, acc[j], false, false);
        }
        __syncthreads();
    }

    // normalize and write ctx (bf16, [token][h*64+v])
    #pragma unroll
    for (int j = 0; j < 4; ++j) {
        int col = h * 64 + j * 16 + l16;
        #pragma unroll
        for (int r = 0; r < 8; ++r) {
            int tok = b * S + q0 + wave * 16 + r + 8 * half;
            Ctxb[(size_t)tok * D + col] = (bf16)(acc[j][r] / lrun[r]);
        }
    }
}

// ---------------------------------------------------------------------------
// LayerNorm: one block per row (D = 512), outputs fp32 and optional bf16
// ---------------------------------------------------------------------------
__global__ __launch_bounds__(256)
void ln_kernel(const float* __restrict__ X, const float* __restrict__ g,
               const float* __restrict__ be, float* __restrict__ outF,
               bf16* __restrict__ outB) {
    __shared__ float red[8];
    const int row = blockIdx.x;
    const int tid = threadIdx.x;
    const int lane = tid & 31, wave = tid >> 5;
    const float* xr = X + (size_t)row * 512;
    float x0 = xr[tid], x1 = xr[tid + 256];

    float s = x0 + x1;
    #pragma unroll
    for (int off = 1; off < 32; off <<= 1) s += __shfl_xor(s, off);
    if (lane == 0) red[wave] = s;
    __syncthreads();
    float mean = 0.f;
    #pragma unroll
    for (int i = 0; i < 8; ++i) mean += red[i];
    mean *= (1.0f / 512.0f);
    __syncthreads();

    float d0 = x0 - mean, d1 = x1 - mean;
    float vs = d0 * d0 + d1 * d1;
    #pragma unroll
    for (int off = 1; off < 32; off <<= 1) vs += __shfl_xor(vs, off);
    if (lane == 0) red[wave] = vs;
    __syncthreads();
    float var = 0.f;
    #pragma unroll
    for (int i = 0; i < 8; ++i) var += red[i];
    var *= (1.0f / 512.0f);
    float inv = rsqrtf(var + 1e-5f);

    float y0 = d0 * inv * g[tid] + be[tid];
    float y1 = d1 * inv * g[tid + 256] + be[tid + 256];
    outF[(size_t)row * 512 + tid] = y0;
    outF[(size_t)row * 512 + tid + 256] = y1;
    if (outB) {
        outB[(size_t)row * 512 + tid] = (bf16)y0;
        outB[(size_t)row * 512 + tid + 256] = (bf16)y1;
    }
}

// ---------------------------------------------------------------------------
extern "C" void kernel_launch(void* const* d_in, const int* in_sizes, int n_in,
                              void* d_out, int out_size, void* d_ws, size_t ws_size,
                              hipStream_t stream) {
    (void)in_sizes; (void)n_in; (void)out_size; (void)ws_size;
    const float* x   = (const float*)d_in[0];
    const float* Wq  = (const float*)d_in[1];
    const float* bq  = (const float*)d_in[2];
    const float* Wk  = (const float*)d_in[3];
    const float* bk  = (const float*)d_in[4];
    const float* Wv  = (const float*)d_in[5];
    const float* bv  = (const float*)d_in[6];
    const float* Wo  = (const float*)d_in[7];
    const float* bo  = (const float*)d_in[8];
    const float* W1  = (const float*)d_in[9];
    const float* b1  = (const float*)d_in[10];
    const float* W2  = (const float*)d_in[11];
    const float* b2  = (const float*)d_in[12];
    const float* g1  = (const float*)d_in[13];
    const float* be1 = (const float*)d_in[14];
    const float* g2  = (const float*)d_in[15];
    const float* be2 = (const float*)d_in[16];

    const int B = 8, S = 1024, D = 512, H = 8, DF = 2048;
    const int M = B * S; // 8192

    char* ws = (char*)d_ws;
    size_t off = 0;
    auto alloc = [&](size_t bytes) {
        void* p = ws + off;
        off = (off + bytes + 255) & ~(size_t)255;
        return p;
    };

    bf16*  xb   = (bf16*)alloc((size_t)M * D * 2);
    bf16*  Wqt  = (bf16*)alloc((size_t)D * D * 2);
    bf16*  Wkt  = (bf16*)alloc((size_t)D * D * 2);
    bf16*  Wvt  = (bf16*)alloc((size_t)D * D * 2);
    bf16*  Wot  = (bf16*)alloc((size_t)D * D * 2);
    bf16*  W1t  = (bf16*)alloc((size_t)D * DF * 2);
    bf16*  W2t  = (bf16*)alloc((size_t)DF * D * 2);
    bf16*  Qb   = (bf16*)alloc((size_t)M * D * 2);
    bf16*  Kbuf = (bf16*)alloc((size_t)M * D * 2);
    bf16*  Vbuf = (bf16*)alloc((size_t)M * D * 2);
    bf16*  Ctxb = (bf16*)alloc((size_t)M * D * 2);
    float* t1   = (float*)alloc((size_t)M * D * 4);
    float* x1   = (float*)alloc((size_t)M * D * 4);
    bf16*  x1b  = (bf16*)alloc((size_t)M * D * 2);
    bf16*  hb   = (bf16*)alloc((size_t)M * DF * 2);
    float* t2   = (float*)alloc((size_t)M * D * 4);

    // --- conversions ---
    cvt_f32_bf16<<<(M * D) / 256, 256, 0, stream>>>(x, xb, M * D);
    cvt_qkv_w<<<(D * D) / 256, 256, 0, stream>>>(Wq, Wqt);
    cvt_qkv_w<<<(D * D) / 256, 256, 0, stream>>>(Wk, Wkt);
    cvt_qkv_w<<<(D * D) / 256, 256, 0, stream>>>(Wv, Wvt);
    transpose_cvt<<<(D * D) / 256, 256, 0, stream>>>(Wo, Wot, D, D);
    transpose_cvt<<<(D * DF) / 256, 256, 0, stream>>>(W1, W1t, D, DF);
    transpose_cvt<<<(DF * D) / 256, 256, 0, stream>>>(W2, W2t, DF, D);

    // --- Q, K, V projections (bias index == flat bq index since col = h*64+k) ---
    dim3 gQKV(D / GBN, M / GBM);
    gemm_bf16<0><<<gQKV, 256, 0, stream>>>(xb, Wqt, bq, nullptr, nullptr, Qb, M, D, D);
    gemm_bf16<0><<<gQKV, 256, 0, stream>>>(xb, Wkt, bk, nullptr, nullptr, Kbuf, M, D, D);
    gemm_bf16<0><<<gQKV, 256, 0, stream>>>(xb, Wvt, bv, nullptr, nullptr, Vbuf, M, D, D);

    // --- attention ---
    attn_kernel<<<dim3(S / 128, H, B), 256, 0, stream>>>(Qb, Kbuf, Vbuf, Ctxb);

    // --- output projection + residual ---
    gemm_bf16<1><<<gQKV, 256, 0, stream>>>(Ctxb, Wot, bo, x, t1, nullptr, M, D, D);
    ln_kernel<<<M, 256, 0, stream>>>(t1, g1, be1, x1, x1b);

    // --- FFN ---
    gemm_bf16<0><<<dim3(DF / GBN, M / GBM), 256, 0, stream>>>(x1b, W1t, b1, nullptr,
                                                              nullptr, hb, M, DF, D);
    gemm_bf16<1><<<gQKV, 256, 0, stream>>>(hb, W2t, b2, x1, t2, nullptr, M, D, DF);
    ln_kernel<<<M, 256, 0, stream>>>(t2, g2, be2, (float*)d_out, nullptr);
}
